// Net_64132451664097
// MI455X (gfx1250) — matrix operand order repaired and checked by reading, not compiled, not run on hardware
//
#include <hip/hip_runtime.h>
#include <hip/hip_bf16.h>
#include <stdint.h>

#define NN      100000
#define NE      1600000
#define CH      64           // hidden/out channels
#define BN_EPS  1e-5f
#define NT_ALL  24           // fused column tiles: lin[0..3] skip[4..7] film[8..15] fskip[16..23]

typedef __attribute__((ext_vector_type(16))) __bf16 v16bf;
typedef __attribute__((ext_vector_type(8)))  float  v8f;

union AFrag { v16bf v; uint32_t u[8]; uint4 q[2]; };

// ---------------- elementwise utilities ----------------
__global__ __launch_bounds__(256) void k_zero(float* __restrict__ p, size_t n) {
    size_t i = (size_t)blockIdx.x * blockDim.x + threadIdx.x;
    size_t s = (size_t)gridDim.x * blockDim.x;
    for (; i < n; i += s) p[i] = 0.0f;
}

__global__ __launch_bounds__(256) void k_cast_bf16(const float* __restrict__ in,
                                                   __bf16* __restrict__ out, size_t n) {
    size_t i = (size_t)blockIdx.x * blockDim.x + threadIdx.x;
    size_t s = (size_t)gridDim.x * blockDim.x;
    for (; i < n; i += s) out[i] = (__bf16)in[i];
}

// ---------------- weight pre-pack into the fused [K x 384] B-fragment image ----------------
// Fragment f = ks*NT_ALL + nt_global, stored [lane][v] contiguous so a wave reads one
// fragment as two coalesced b128 loads. Contents (contiguous K order, same as the A loads):
//   lane: col = nt*16 + (lane&15); half = lane>>4 selects K 0..15 / 16..31 of the step;
//   dword v packs bf16 of K = ks*32 + half*16 + 2v (lo) and +1 (hi).
__global__ __launch_bounds__(256) void k_pack_w(const float* __restrict__ W,
                                                uint32_t* __restrict__ Wp,
                                                int K, int nc, int nt_off) {
    int total = (K * nc) >> 1;                 // packed dwords for this matrix
    int id = blockIdx.x * blockDim.x + threadIdx.x;
    if (id >= total) return;
    int ntl  = nc >> 4;
    int v    = id & 7;
    int lane = (id >> 3) & 31;
    int fl   = id >> 8;
    int nt   = fl % ntl;
    int ks   = fl / ntl;
    int col  = nt * 16 + (lane & 15);
    int k    = ks * 32 + (lane >> 4) * 16 + 2 * v;
    union { __bf16 b; uint16_t u; } lo, hi;
    lo.b = (__bf16)W[(size_t)k * nc + col];
    hi.b = (__bf16)W[(size_t)(k + 1) * nc + col];
    Wp[(size_t)(((ks * NT_ALL + nt_off + nt) * 32 + lane) * 8 + v)] =
        (uint32_t)lo.u | ((uint32_t)hi.u << 16);
}

// ---------------- fused per-layer node GEMM ----------------
// One wave owns a 16-row M tile and ALL 24 column tiles of the fused weight image:
// each A fragment (2 x b128) feeds 24 v_wmma issues. Epilogue writes lin and film
// (+bias) and computes the skip-branch FiLM  relu(gamma_s*skip + beta_s)  entirely
// in registers (fskip never touches memory).
__global__ __launch_bounds__(256) void k_layer_gemm(
    const __bf16* __restrict__ X, const uint4* __restrict__ Wp,
    const float* __restrict__ film_b, const float* __restrict__ fskip_b,
    float* __restrict__ lin, float* __restrict__ film, float* __restrict__ skipo,
    int M, int K, int act)
{
    const int lane  = threadIdx.x & 31;
    const int wave  = threadIdx.x >> 5;
    const int mtile = blockIdx.x * 8 + wave;
    if (mtile * 16 >= M) return;
    const int half = lane >> 4;     // 0: lanes 0-15, 1: lanes 16-31
    const int lr   = lane & 15;     // row-in-tile (A) / col-in-tile (B,C,D)
    const int row0 = mtile * 16;

    const __bf16* xrow = X + (size_t)(row0 + lr) * K;
    v8f acc[NT_ALL] = {};
    const int KS = K >> 5;
    for (int ks = 0; ks < KS; ++ks) {
        AFrag a;
        // 16-bit A 16x32 layout (matches packed B): lanes0-15 K{0..15}, lanes16-31 K{16..31}
        a.q[0] = *(const uint4*)(xrow + ks * 32 + half * 16);
        a.q[1] = *(const uint4*)(xrow + ks * 32 + half * 16 + 8);
        const uint4* wq = Wp + ((size_t)(ks * NT_ALL) * 32 + lane) * 2;
        #pragma unroll
        for (int nt = 0; nt < NT_ALL; ++nt) {
            AFrag b;
            b.q[0] = wq[0];
            b.q[1] = wq[1];
            acc[nt] = __builtin_amdgcn_wmma_f32_16x16x32_bf16(
                false, a.v, false, b.v, (short)0, acc[nt], false, false);
            wq += 64;               // next fragment: 32 lanes * 2 uint4
        }
    }

    // ---- epilogue (f32 C/D layout: VGPR r -> row r + half*8) ----
    // lin: tiles 0..3, no bias
    #pragma unroll
    for (int nt = 0; nt < 4; ++nt) {
        float* yp = lin + (size_t)row0 * CH + nt * 16 + lr;
        #pragma unroll
        for (int r = 0; r < 8; ++r)
            yp[(size_t)(r + half * 8) * CH] = acc[nt][r];
    }
    // film: tiles 8..15, bias film_b
    #pragma unroll
    for (int nt = 0; nt < 8; ++nt) {
        const float bv = film_b[nt * 16 + lr];
        float* yp = film + (size_t)row0 * 2 * CH + nt * 16 + lr;
        #pragma unroll
        for (int r = 0; r < 8; ++r)
            yp[(size_t)(r + half * 8) * 2 * CH] = acc[8 + nt][r] + bv;
    }
    // skip-branch FiLM in registers: skip tiles 4..7; fskip beta 16..19, gamma 20..23
    #pragma unroll
    for (int nt = 0; nt < 4; ++nt) {
        const float bb = fskip_b[nt * 16 + lr];        // beta bias  (cols 0..63)
        const float bg = fskip_b[CH + nt * 16 + lr];   // gamma bias (cols 64..127)
        float* yp = skipo + (size_t)row0 * CH + nt * 16 + lr;
        #pragma unroll
        for (int r = 0; r < 8; ++r) {
            float beta  = acc[16 + nt][r] + bb;
            float gamma = acc[20 + nt][r] + bg;
            float v = fmaf(gamma, acc[4 + nt][r], beta);
            if (act) v = fmaxf(v, 0.0f);
            yp[(size_t)(r + half * 8) * CH] = v;
        }
    }
}

// ---------------- degree (computed once, reused by all 3 layers) ----------------
__global__ __launch_bounds__(256) void k_degree(const int* __restrict__ dst,
                                                float* __restrict__ cnt) {
    int e = blockIdx.x * blockDim.x + threadIdx.x;
    if (e < NE) atomicAdd(&cnt[dst[e]], 1.0f);
}

// ---------------- edge FiLM message + scatter-add (L2-resident tables) ----------------
// 16 threads per edge, each handles 4 channels with float4 gathers + 4 f32 atomics.
__global__ __launch_bounds__(256) void k_edge(const int* __restrict__ src,
                                              const int* __restrict__ dst,
                                              const float* __restrict__ lin,
                                              const float* __restrict__ film,
                                              float* __restrict__ agg, int act)
{
    size_t t = (size_t)blockIdx.x * blockDim.x + threadIdx.x;
    size_t e = t >> 4;
    if (e >= (size_t)NE) return;
    int c4 = (int)(t & 15) << 2;
    int s = src[e], d = dst[e];
    const float4 xj = *(const float4*)(lin  + (size_t)s * CH + c4);
    const float4 be = *(const float4*)(film + (size_t)d * 2 * CH + c4);
    const float4 ga = *(const float4*)(film + (size_t)d * 2 * CH + CH + c4);
    float4 m;
    m.x = fmaf(ga.x, xj.x, be.x);
    m.y = fmaf(ga.y, xj.y, be.y);
    m.z = fmaf(ga.z, xj.z, be.z);
    m.w = fmaf(ga.w, xj.w, be.w);
    if (act) {
        m.x = fmaxf(m.x, 0.0f); m.y = fmaxf(m.y, 0.0f);
        m.z = fmaxf(m.z, 0.0f); m.w = fmaxf(m.w, 0.0f);
    }
    float* ap = agg + (size_t)d * CH + c4;
    atomicAdd(ap + 0, m.x);
    atomicAdd(ap + 1, m.y);
    atomicAdd(ap + 2, m.z);
    atomicAdd(ap + 3, m.w);
}

// ---------------- combine: h = skip_out + agg / max(deg,1) ----------------
__global__ __launch_bounds__(256) void k_combine(const float* __restrict__ skipo,
                                                 const float* __restrict__ agg,
                                                 const float* __restrict__ cnt,
                                                 float* __restrict__ h) {
    size_t i = (size_t)blockIdx.x * blockDim.x + threadIdx.x;
    if (i >= (size_t)NN * CH) return;
    size_t n = i >> 6;
    float inv = 1.0f / fmaxf(cnt[n], 1.0f);
    h[i] = fmaf(agg[i], inv, skipo[i]);
}

// ---------------- BatchNorm: per-channel sum/sumsq then normalize ----------------
__global__ __launch_bounds__(256) void k_bn_reduce(const float* __restrict__ h,
                                                   float* __restrict__ stats) {
    int c  = threadIdx.x & (CH - 1);
    int rl = threadIdx.x >> 6;               // 4 row-lanes x 64 channels
    float s = 0.0f, s2 = 0.0f;
    for (int n = blockIdx.x * 4 + rl; n < NN; n += gridDim.x * 4) {
        float v = h[(size_t)n * CH + c];
        s += v; s2 += v * v;
    }
    __shared__ float ls[256], ls2[256];
    ls[threadIdx.x] = s; ls2[threadIdx.x] = s2;
    __syncthreads();
    if (rl == 0) {
        s  = ls [c] + ls [CH + c] + ls [2 * CH + c] + ls [3 * CH + c];
        s2 = ls2[c] + ls2[CH + c] + ls2[2 * CH + c] + ls2[3 * CH + c];
        atomicAdd(&stats[c], s);
        atomicAdd(&stats[CH + c], s2);
    }
}

__global__ __launch_bounds__(256) void k_bn_apply(float* __restrict__ h,
                                                  const float* __restrict__ stats,
                                                  const float* __restrict__ g,
                                                  const float* __restrict__ b) {
    size_t i = (size_t)blockIdx.x * blockDim.x + threadIdx.x;
    if (i >= (size_t)NN * CH) return;
    int c = (int)(i & 63);
    float mean = stats[c] * (1.0f / NN);
    float var  = stats[CH + c] * (1.0f / NN) - mean * mean;
    h[i] = (h[i] - mean) * rsqrtf(var + BN_EPS) * g[c] + b[c];
}

// =====================================================================
extern "C" void kernel_launch(void* const* d_in, const int* in_sizes, int n_in,
                              void* d_out, int out_size, void* d_ws, size_t ws_size,
                              hipStream_t stream) {
    const float* x   = (const float*)d_in[0];
    const int*   ei  = (const int*)d_in[1];
    const int*   src = ei;
    const int*   dst = ei + NE;

    // ---- carve workspace (256B-aligned regions) ----
    char* wp_ = (char*)d_ws;
    auto carve = [&](size_t bytes) -> void* {
        void* r = (void*)wp_;
        wp_ += (bytes + 255) & ~(size_t)255;
        return r;
    };
    __bf16* xb    = (__bf16*)carve((size_t)NN * 128 * sizeof(__bf16)); // 25.6 MB
    float*  lin   = (float*) carve((size_t)NN * CH * 4);               // 25.6 MB
    float*  film  = (float*) carve((size_t)NN * 2 * CH * 4);           // 51.2 MB
    float*  skipo = (float*) carve((size_t)NN * CH * 4);               // 25.6 MB
    float*  agg   = (float*) carve((size_t)NN * CH * 4);               // 25.6 MB
    float*  h     = (float*) carve((size_t)NN * CH * 4);               // 25.6 MB
    float*  cnt   = (float*) carve((size_t)NN * 4);
    float*  stats = (float*) carve(2 * CH * 4);

    // ---- fused packed weight image per layer: [K x 384], tiles lin|skip|film|fskip ----
    // input indices per layer: lin, film, skip, fskip
    const int widx[3][4] = { {2, 3, 5, 6}, {10, 11, 13, 14}, {18, 19, 21, 22} };
    const int kdim[3]    = { 128, 64, 64 };
    uint4* wfused[3];
    for (int l = 0; l < 3; ++l) {
        int K = kdim[l];
        wfused[l] = (uint4*)carve((size_t)K * 16 * NT_ALL * 2);   // K*384 bf16
        struct { int idx, nc, off; } parts[4] = {
            { widx[l][0],  64,  0 },    // lin   -> tiles 0..3
            { widx[l][2],  64,  4 },    // skip  -> tiles 4..7
            { widx[l][1], 128,  8 },    // film  -> tiles 8..15
            { widx[l][3], 128, 16 },    // fskip -> tiles 16..23
        };
        for (int p = 0; p < 4; ++p) {
            int ndw = (K * parts[p].nc) >> 1;
            k_pack_w<<<(ndw + 255) / 256, 256, 0, stream>>>(
                (const float*)d_in[parts[p].idx], (uint32_t*)wfused[l],
                K, parts[p].nc, parts[p].off);
        }
    }

    const int MT      = NN / 16;                        // 6250 exact M tiles
    const int EW_GRID = (int)(((size_t)NN * CH + 255) / 256);
    const int ED_GRID = (int)(((size_t)NE * 16 + 255) / 256);
    const int GG      = (MT + 7) / 8;                   // gemm grid (8 waves/block)

    // ---- destination degrees (identical for all layers) ----
    k_zero<<<256, 256, 0, stream>>>(cnt, NN);
    k_degree<<<(NE + 255) / 256, 256, 0, stream>>>(dst, cnt);

    auto run_layer = [&](const float* xin, int l,
                         const float* bfilm, const float* bfskip,
                         float* hout, int act) {
        int K = kdim[l];
        k_cast_bf16<<<8192, 256, 0, stream>>>(xin, xb, (size_t)NN * K);
        k_layer_gemm<<<GG, 256, 0, stream>>>(xb, wfused[l], bfilm, bfskip,
                                             lin, film, skipo, NN, K, act);
        k_zero<<<4096, 256, 0, stream>>>(agg, (size_t)NN * CH);
        k_edge<<<ED_GRID, 256, 0, stream>>>(src, dst, lin, film, agg, act);
        k_combine<<<EW_GRID, 256, 0, stream>>>(skipo, agg, cnt, hout);
    };
    auto run_bn = [&](float* hbuf, const float* g, const float* b) {
        k_zero<<<1, 128, 0, stream>>>(stats, 2 * CH);
        k_bn_reduce<<<256, 256, 0, stream>>>(hbuf, stats);
        k_bn_apply<<<EW_GRID, 256, 0, stream>>>(hbuf, stats, g, b);
    };

    // layer 0: 128 -> 64, relu, BN
    run_layer(x, 0, (const float*)d_in[4], (const float*)d_in[7], h, 1);
    run_bn(h, (const float*)d_in[8], (const float*)d_in[9]);

    // layer 1: 64 -> 64, relu, BN   (h is consumed by the cast before it is rewritten)
    run_layer(h, 1, (const float*)d_in[12], (const float*)d_in[15], h, 1);
    run_bn(h, (const float*)d_in[16], (const float*)d_in[17]);

    // layer 2: 64 -> 64, no act, no BN, straight into d_out
    run_layer(h, 2, (const float*)d_in[20], (const float*)d_in[23], (float*)d_out, 0);

    (void)in_sizes; (void)n_in; (void)out_size; (void)ws_size;
}
